// EncoderLayer_26723286515881
// MI455X (gfx1250) — matrix operand order repaired and checked by reading, not compiled, not run on hardware
//
#include <hip/hip_runtime.h>

// ---------------------------------------------------------------------------
// CDNA5 (gfx1250) wave32 WMMA transformer encoder layer.
// bf16 matrix cores (v_wmma_f32_16x16x32_bf16), f32 accumulate.
// All attention operand movement (Q, K, V tiles) uses async global->LDS DMA
// (global_load_async_to_lds_b128 / s_wait_asynccnt), overlapped with WMMA.
// V is produced pre-transposed by its projection GEMM so no runtime transpose.
// ---------------------------------------------------------------------------

typedef __attribute__((ext_vector_type(16))) __bf16 v16bf;
typedef __attribute__((ext_vector_type(8)))  float  v8f;

union AFrag { v16bf v; unsigned u[8]; };
union CFrag { v8f   v; float    f[8]; };

__device__ inline unsigned short f32_to_bf16(float f) {
    unsigned u = __builtin_bit_cast(unsigned, f);
    u += 0x7fffu + ((u >> 16) & 1u);          // round-to-nearest-even
    return (unsigned short)(u >> 16);
}
__device__ inline unsigned pack_bf16(float lo, float hi) {
    return (unsigned)f32_to_bf16(lo) | ((unsigned)f32_to_bf16(hi) << 16);
}

// Async DMA copy of 16 bytes global -> LDS (ASYNCcnt tracked).
__device__ inline void async_copy_b128(unsigned ldsOff, const void* gaddr) {
#if defined(__gfx1250__)
    asm volatile("global_load_async_to_lds_b128 %0, %1, off"
                 :: "v"(ldsOff), "v"(gaddr) : "memory");
#endif
}
__device__ inline void wait_async0() {
#if defined(__gfx1250__)
    asm volatile("s_wait_asynccnt 0x0" ::: "memory");
#endif
}

// A-matrix 16x32 bf16 fragment (ISA 7.12.2): lane<16 -> K=0..7,16..23 ;
// lane>=16 -> K=8..15,24..31. LDS is row-major packed-uint pairs.
__device__ inline void load_frag_a(AFrag& f, const unsigned* lds,
                                   int rowBase, int kBaseU, int strideU, int lane) {
    const unsigned* p = lds + (rowBase + (lane & 15)) * strideU + kBaseU
                            + ((lane >> 4) << 2);
#pragma unroll
    for (int q = 0; q < 8; ++q)
        f.u[q] = p[(q & 3) + ((q >> 2) << 3)];
}

// B-matrix 32x16 bf16 fragment: lane = column N, lanes 0-15 K=0..7,16..23,
// lanes 16-31 K=8..15,24..31. LDS holds B transposed: [n][k-pair].
__device__ inline void load_frag_b(AFrag& f, const unsigned* lds,
                                   int rowBase, int kBaseU, int strideU, int lane) {
    const unsigned* p = lds + (rowBase + (lane & 15)) * strideU + kBaseU
                            + ((lane >> 4) << 2);
#pragma unroll
    for (int q = 0; q < 8; ++q)
        f.u[q] = p[(q & 3) + ((q >> 2) << 3)];
}

__device__ inline v8f wmma_bf16(const AFrag& a, const AFrag& b, v8f c) {
    return __builtin_amdgcn_wmma_f32_16x16x32_bf16(
        false, a.v, false, b.v, (short)0, c, false, false);
}

// ---------------------------------------------------------------------------
// Generic WMMA GEMM: out[M,N] = epilogue(A[M,K] @ W[K,N]).
// Block tile 128x64, 8 waves, 4 C-tiles per wave (2m x 2n), K step 32.
// Double-buffered LDS, software-pipelined global staging.
// OUTT=1: bf16 output stored transposed: out[col*ldT + row].
// ---------------------------------------------------------------------------
template<int ABF16, int OUTBF16, int OUTT, int RELU, int BIAS, int RESID>
__global__ __launch_bounds__(256) void gemm_wmma_kernel(
    const void* __restrict__ Ap, const float* __restrict__ Bw,
    const float* __restrict__ bias, const float* __restrict__ resid,
    void* __restrict__ outp, int M, int N, int K, int ldT, float out_scale)
{
    __shared__ __align__(16) unsigned Asl[2][128 * 18];  // [row][k-pair]
    __shared__ __align__(16) unsigned Bsl[2][64 * 18];   // [n][k-pair] (B transposed)

    const int tid   = threadIdx.x;
    const int lane  = tid & 31;
    const int w     = tid >> 5;
    const int mBase = blockIdx.y * 128;
    const int nBase = blockIdx.x * 64;
    const int mi0   = (w & 3) << 1;    // 0,2,4,6
    const int ni0   = (w >> 2) << 1;   // 0,2

    CFrag acc[2][2];
#pragma unroll
    for (int i = 0; i < 8; ++i) {
        acc[0][0].f[i] = 0.f; acc[0][1].f[i] = 0.f;
        acc[1][0].f[i] = 0.f; acc[1][1].f[i] = 0.f;
    }

    // --- hoisted per-thread staging bases ---
    const int aRow = tid >> 4, aCol = tid & 15;
    const float*    AfBase  = (const float*)Ap    + (size_t)(mBase + aRow) * K + 2 * aCol;
    const unsigned* A32Base = (const unsigned*)Ap + (size_t)(mBase + aRow) * (K >> 1) + aCol;
    const int aLds = aRow * 18 + aCol;
    const int bN = tid & 63, bK = tid >> 6;
    const float* BwBase = Bw + (size_t)(2 * bK) * N + nBase + bN;
    const int bLds = bN * 18 + bK;

    unsigned ta[8], tb[4];
    auto loadStage = [&](int ko) {
        if (ABF16) {
            const unsigned* ap = A32Base + (ko >> 1);
#pragma unroll
            for (int p = 0; p < 8; ++p)
                ta[p] = ap[(size_t)p * 16 * (K >> 1)];
        } else {
            const float* ap = AfBase + ko;
#pragma unroll
            for (int p = 0; p < 8; ++p) {
                float2 f2 = *(const float2*)(ap + (size_t)p * 16 * K);
                ta[p] = pack_bf16(f2.x, f2.y);
            }
        }
        const float* bp = BwBase + (size_t)ko * N;
#pragma unroll
        for (int p = 0; p < 4; ++p) {
            const float* s = bp + (size_t)p * 8 * N;
            tb[p] = pack_bf16(s[0], s[N]);
        }
    };

    loadStage(0);
    for (int ko = 0; ko < K; ko += 32) {
        unsigned* As = Asl[(ko >> 5) & 1];
        unsigned* Bs = Bsl[(ko >> 5) & 1];
#pragma unroll
        for (int p = 0; p < 8; ++p) As[aLds + p * 16 * 18] = ta[p];
#pragma unroll
        for (int p = 0; p < 4; ++p) Bs[bLds + p * 4] = tb[p];
        __syncthreads();

        if (ko + 32 < K) {
            loadStage(ko + 32);                    // latency hidden under WMMA
            __builtin_prefetch(BwBase + (size_t)(ko + 64) * N, 0, 1);
        }

        AFrag a0, a1, b0, b1;
        load_frag_a(a0, As, (mi0)     * 16, 0, 18, lane);
        load_frag_a(a1, As, (mi0 + 1) * 16, 0, 18, lane);
        load_frag_b(b0, Bs, (ni0)     * 16, 0, 18, lane);
        load_frag_b(b1, Bs, (ni0 + 1) * 16, 0, 18, lane);
        acc[0][0].v = wmma_bf16(a0, b0, acc[0][0].v);
        acc[0][1].v = wmma_bf16(a0, b1, acc[0][1].v);
        acc[1][0].v = wmma_bf16(a1, b0, acc[1][0].v);
        acc[1][1].v = wmma_bf16(a1, b1, acc[1][1].v);
        // no trailing barrier: next iteration writes the other LDS buffer
    }

    // --- epilogue ---
    const int hi8 = (lane >> 4) << 3;
    const int nc  = lane & 15;
#pragma unroll
    for (int ti = 0; ti < 2; ++ti) {
#pragma unroll
        for (int tj = 0; tj < 2; ++tj) {
#pragma unroll
            for (int i = 0; i < 8; ++i) {
                int row = mBase + (mi0 + ti) * 16 + i + hi8;
                int col = nBase + (ni0 + tj) * 16 + nc;
                float v = acc[ti][tj].f[i] * out_scale;
                if (BIAS)  v += bias[col];
                if (RESID) v += resid[(size_t)row * N + col];
                if (RELU)  v = fmaxf(v, 0.f);
                if (OUTBF16) {
                    size_t idx = OUTT ? ((size_t)col * ldT + row)
                                      : ((size_t)row * N + col);
                    ((unsigned short*)outp)[idx] = f32_to_bf16(v);
                } else {
                    ((float*)outp)[(size_t)row * N + col] = v;
                }
            }
        }
    }
}

// ---------------------------------------------------------------------------
// Flash attention: one block = 64 query rows of one (b,h).
// Q pre-scaled by 1/sqrt(dk). Online softmax, QK^T and PV via WMMA.
// Q, K and (pre-transposed) V tiles all staged via async global->LDS DMA.
// Next key block's K+V DMA issued after the scores barrier, overlapping
// softmax + PV. V double-buffered so its DMA can overlap the PV reads.
// ---------------------------------------------------------------------------
__global__ __launch_bounds__(256) void attn_kernel(
    const unsigned short* __restrict__ Qb, const unsigned short* __restrict__ Kb,
    const unsigned short* __restrict__ Vt,   // [D][BS] transposed values
    const int* __restrict__ mask,
    unsigned short* __restrict__ ctxb, int S, int D, int H, int BS)
{
    __shared__ __align__(16) unsigned QT[64 * 36];      // [qrow][dk-pair]
    __shared__ __align__(16) unsigned KT[64 * 36];      // [keyrow][dk-pair]
    __shared__ __align__(16) unsigned VT[2][64 * 36];   // [dkcol][key-pair]
    __shared__ __align__(16) unsigned PT[64 * 34];      // [qrow][key-pair]
    __shared__ float SC[64 * 65];                       // scores
    __shared__ float mrun[64], lrun[64], corr[64];

    const int tid  = threadIdx.x;
    const int lane = tid & 31;
    const int w    = tid >> 5;
    const int qb   = blockIdx.x;
    const int b    = blockIdx.y / H;
    const int h    = blockIdx.y % H;
    const int dk   = D / H;                 // 64
    const int DU   = D >> 1;                // Q/K row stride in uints
    const int BSU  = BS >> 1;               // Vt row stride in uints
    const int nkb  = S / 64;

    // async copy mapping: 512 b128 chunks per 64x64 bf16 tile, 2 per thread:
    //   r = p*32 + tid>>3 (tile row), c4 = (tid&7)*4 (uint column)
    const int cr = tid >> 3, cc4 = (tid & 7) << 2;
    const unsigned pStrideL = 32 * 36 * 4;              // p=1 LDS offset

    const unsigned qLds0 = (unsigned)(size_t)&QT[cr * 36 + cc4];
    const unsigned kLds0 = (unsigned)(size_t)&KT[cr * 36 + cc4];
    const unsigned vLds0[2] = { (unsigned)(size_t)&VT[0][cr * 36 + cc4],
                                (unsigned)(size_t)&VT[1][cr * 36 + cc4] };

    // stage Q tile (async DMA)
    {
        const char* g = (const char*)((const unsigned*)Qb
                      + (size_t)(b * S + qb * 64 + cr) * DU + h * 32 + cc4);
#pragma unroll
        for (int p = 0; p < 2; ++p)
            async_copy_b128(qLds0 + p * pStrideL, g + (size_t)p * 32 * DU * 4);
    }
    if (tid < 64) { mrun[tid] = -3.0e38f; lrun[tid] = 0.f; }

    // K: rows = keys (stride DU uints); advance 64 rows per block
    const char* kPtr = (const char*)((const unsigned*)Kb
                     + (size_t)(b * S + cr) * DU + h * 32 + cc4);
    auto issueK = [&]() {
#pragma unroll
        for (int p = 0; p < 2; ++p)
            async_copy_b128(kLds0 + p * pStrideL, kPtr + (size_t)p * 32 * DU * 4);
        kPtr += (size_t)64 * DU * 4;
    };
    // V (transposed): rows = dk (stride BSU uints); advance 64 keys = 128 B
    const char* vPtr = (const char*)((const unsigned*)Vt
                     + (size_t)(h * 64 + cr) * BSU + ((b * S) >> 1) + cc4);
    auto issueV = [&](int buf) {
#pragma unroll
        for (int p = 0; p < 2; ++p)
            async_copy_b128(vLds0[buf] + p * pStrideL, vPtr + (size_t)p * 32 * BSU * 4);
        vPtr += 128;
    };
    issueK();
    issueV(0);

    const int* mPtr = mask + b * S;
    const int mi  = w >> 1;
    const int ni0 = (w & 1) << 1;
    const int hi8 = (lane >> 4) << 3;
    const int nc  = lane & 15;

    CFrag acc0, acc1;
#pragma unroll
    for (int i = 0; i < 8; ++i) { acc0.f[i] = 0.f; acc1.f[i] = 0.f; }

    for (int kb = 0; kb < nkb; ++kb) {
        wait_async0();                  // Q (first iter) + K(kb) + V(kb) in LDS
        __syncthreads();

        // scores = Q @ K^T (dk=64 -> two K=32 WMMA steps)
        CFrag s0, s1;
#pragma unroll
        for (int i = 0; i < 8; ++i) { s0.f[i] = 0.f; s1.f[i] = 0.f; }
#pragma unroll
        for (int kk = 0; kk < 2; ++kk) {
            AFrag a, b0, b1;
            load_frag_a(a,  QT, mi  * 16,       kk * 16, 36, lane);
            load_frag_b(b0, KT, ni0 * 16,       kk * 16, 36, lane);
            load_frag_b(b1, KT, (ni0 + 1) * 16, kk * 16, 36, lane);
            s0.v = wmma_bf16(a, b0, s0.v);
            s1.v = wmma_bf16(a, b1, s1.v);
        }
#pragma unroll
        for (int i = 0; i < 8; ++i) {
            int row = mi * 16 + i + hi8;
            int c0  = ni0 * 16 + nc, c1 = c0 + 16;
            SC[row * 65 + c0] = mPtr[c0] ? s0.f[i] : -1e9f;
            SC[row * 65 + c1] = mPtr[c1] ? s1.f[i] : -1e9f;
        }
        __syncthreads();                // KT consumption complete

        if (kb + 1 < nkb) {             // next K+V DMA overlaps softmax + PV
            issueK();
            issueV((kb + 1) & 1);
        }

        // online softmax: one thread per query row
        if (tid < 64) {
            int j = tid;
            float mx = mrun[j];
            const float* sc = &SC[j * 65];
            float bm = -3.0e38f;
#pragma unroll 8
            for (int i2 = 0; i2 < 64; ++i2) bm = fmaxf(bm, sc[i2]);
            float nm = fmaxf(mx, bm);
            float cf = __expf(mx - nm);
            float sum = 0.f;
            unsigned* pt = &PT[j * 34];
#pragma unroll 8
            for (int i2 = 0; i2 < 64; i2 += 2) {
                float p0 = __expf(sc[i2]     - nm);
                float p1 = __expf(sc[i2 + 1] - nm);
                sum += p0 + p1;
                pt[i2 >> 1] = pack_bf16(p0, p1);
            }
            lrun[j] = lrun[j] * cf + sum;
            mrun[j] = nm;
            corr[j] = cf;
        }
        __syncthreads();

        // rescale running context, then ctx += P @ V
        const unsigned* Vb = VT[kb & 1];
#pragma unroll
        for (int i = 0; i < 8; ++i) {
            float cf = corr[mi * 16 + i + hi8];
            acc0.f[i] *= cf; acc1.f[i] *= cf;
        }
#pragma unroll
        for (int kk = 0; kk < 2; ++kk) {
            AFrag a, b0, b1;
            load_frag_a(a,  PT, mi  * 16,       kk * 16, 34, lane);
            load_frag_b(b0, Vb, ni0 * 16,       kk * 16, 36, lane);
            load_frag_b(b1, Vb, (ni0 + 1) * 16, kk * 16, 36, lane);
            acc0.v = wmma_bf16(a, b0, acc0.v);
            acc1.v = wmma_bf16(a, b1, acc1.v);
        }
        __syncthreads();                // PT free for next iteration

        mPtr += 64;
    }

    // normalize and write ctx (bf16, [BS, D] with head offset)
#pragma unroll
    for (int i = 0; i < 8; ++i) {
        int row = mi * 16 + i + hi8;
        float inv = 1.f / lrun[row];
        int c0 = ni0 * 16 + nc, c1 = c0 + 16;
        size_t base = (size_t)(b * S + qb * 64 + row) * D + h * dk;
        ctxb[base + c0] = f32_to_bf16(acc0.f[i] * inv);
        ctxb[base + c1] = f32_to_bf16(acc1.f[i] * inv);
    }
}

// ---------------------------------------------------------------------------
// Row LayerNorm: one block per row of D=1024, 256 threads x 4 elements.
// ---------------------------------------------------------------------------
__global__ __launch_bounds__(256) void ln_kernel(
    const float* __restrict__ in, const float* __restrict__ g,
    const float* __restrict__ be, float* __restrict__ out, int D)
{
    __shared__ float red[256];
    const int tid = threadIdx.x;
    const size_t row = blockIdx.x;
    const float* x = in + row * D;

    float v[4];
    float s = 0.f;
#pragma unroll
    for (int i = 0; i < 4; ++i) { v[i] = x[tid + i * 256]; s += v[i]; }
    red[tid] = s; __syncthreads();
    for (int o = 128; o > 0; o >>= 1) { if (tid < o) red[tid] += red[tid + o]; __syncthreads(); }
    float mu = red[0] / D;
    __syncthreads();

    float q = 0.f;
#pragma unroll
    for (int i = 0; i < 4; ++i) { float d = v[i] - mu; q += d * d; }
    red[tid] = q; __syncthreads();
    for (int o = 128; o > 0; o >>= 1) { if (tid < o) red[tid] += red[tid + o]; __syncthreads(); }
    float rstd = rsqrtf(red[0] / D + 1e-5f);

    float* y = out + row * D;
#pragma unroll
    for (int i = 0; i < 4; ++i) {
        int c = tid + i * 256;
        y[c] = (v[i] - mu) * rstd * g[c] + be[c];
    }
}

// ---------------------------------------------------------------------------
extern "C" void kernel_launch(void* const* d_in, const int* in_sizes, int n_in,
                              void* d_out, int out_size, void* d_ws, size_t ws_size,
                              hipStream_t stream)
{
    (void)in_sizes; (void)n_in; (void)out_size; (void)ws_size;

    const float* x    = (const float*)d_in[0];
    const int*   mask = (const int*)  d_in[1];
    const float* Wq   = (const float*)d_in[2];
    const float* Wk   = (const float*)d_in[3];
    const float* Wv   = (const float*)d_in[4];
    const float* Wo   = (const float*)d_in[5];
    const float* W1   = (const float*)d_in[6];
    const float* b1   = (const float*)d_in[7];
    const float* W2   = (const float*)d_in[8];
    const float* b2   = (const float*)d_in[9];
    const float* ln1g = (const float*)d_in[10];
    const float* ln1b = (const float*)d_in[11];
    const float* ln2g = (const float*)d_in[12];
    const float* ln2b = (const float*)d_in[13];

    const int B = 4, S = 2048, D = 1024, H = 16, Dff = 4096;
    const int BS = B * S;

    // workspace layout (bytes)
    char* ws = (char*)d_ws;
    unsigned short* Qw   = (unsigned short*)(ws);                         // 16 MB bf16
    unsigned short* Kw   = (unsigned short*)(ws + ((size_t)16  << 20));   // 16 MB
    unsigned short* Vtw  = (unsigned short*)(ws + ((size_t)32  << 20));   // 16 MB [D][BS]
    unsigned short* ctxw = (unsigned short*)(ws + ((size_t)48  << 20));   // 16 MB
    float*          y1   = (float*)         (ws + ((size_t)64  << 20));   // 32 MB f32
    float*          hbuf = (float*)         (ws + ((size_t)96  << 20));   // 32 MB f32
    unsigned short* ffw  = (unsigned short*)(ws + ((size_t)128 << 20));   // 64 MB bf16
    float*          y2   = (float*)         (ws + ((size_t)192 << 20));   // 32 MB f32

    dim3 blk(256);
    dim3 gD(D / 64, BS / 128);       // 16 x 64
    dim3 gFF(Dff / 64, BS / 128);    // 64 x 64

    // Q/K/V projections (1/sqrt(dk)=0.125 folded into Q; V stored transposed)
    gemm_wmma_kernel<0,1,0,0,0,0><<<gD, blk, 0, stream>>>(x, Wq, nullptr, nullptr, Qw,  BS, D, D, 0,  0.125f);
    gemm_wmma_kernel<0,1,0,0,0,0><<<gD, blk, 0, stream>>>(x, Wk, nullptr, nullptr, Kw,  BS, D, D, 0,  1.0f);
    gemm_wmma_kernel<0,1,1,0,0,0><<<gD, blk, 0, stream>>>(x, Wv, nullptr, nullptr, Vtw, BS, D, D, BS, 1.0f);

    // flash attention -> ctx
    attn_kernel<<<dim3(S / 64, B * H), blk, 0, stream>>>(Qw, Kw, Vtw, mask, ctxw, S, D, H, BS);

    // out projection + residual, LN1
    gemm_wmma_kernel<1,0,0,0,0,1><<<gD, blk, 0, stream>>>(ctxw, Wo, nullptr, x, y1, BS, D, D, 0, 1.0f);
    ln_kernel<<<BS, blk, 0, stream>>>(y1, ln1g, ln1b, hbuf, D);

    // FFN
    gemm_wmma_kernel<0,1,0,1,1,0><<<gFF, blk, 0, stream>>>(hbuf, W1, b1, nullptr, ffw, BS, Dff, D, 0, 1.0f);
    gemm_wmma_kernel<1,0,0,0,1,1><<<gD,  blk, 0, stream>>>(ffw,  W2, b2, hbuf,   y2, BS, D, Dff, 0, 1.0f);
    ln_kernel<<<BS, blk, 0, stream>>>(y2, ln2g, ln2b, (float*)d_out, D);
}
